// TransformerLanguageModel_45629732553165
// MI455X (gfx1250) — matrix operand order repaired and checked by reading, not compiled, not run on hardware
//
#include <hip/hip_runtime.h>

// ---------------------------------------------------------------------------
// CDNA5 (gfx1250, wave32) transformer LM forward: embed -> QKV -> attention
// -> LM head -> log-softmax NLL.  All GEMMs use v_wmma_f32_16x16x32_bf16.
// Wave tiling: 2 M-tiles x 4 N-tiles per wave in the dense GEMMs (B-fragment
// reuse across M halves the load-issue per WMMA); causal skipping in attn.
// ---------------------------------------------------------------------------

typedef __attribute__((ext_vector_type(16))) __bf16 v16bf;
typedef __attribute__((ext_vector_type(8)))  float  v8f;
typedef __attribute__((ext_vector_type(4)))  float  f32x4;

#define BATCH   4
#define SEQ     2048
#define EMB     1024
#define HEAD    1024
#define VOCAB   50257
#define MROWS   (BATCH * SEQ)            // 8192
#define KT_E    (EMB / 32)               // 32 k-tiles of 32
#define NT_LM_PAD 3144                   // ceil(50257/16)=3142, padded to %4==0

#define WMMA_BF16(a, b, c) \
  __builtin_amdgcn_wmma_f32_16x16x32_bf16(false, (a), false, (b), (short)0, (c), false, false)

static __device__ __forceinline__ v16bf frag_from2(f32x4 lo, f32x4 hi) {
  union { f32x4 f[2]; v16bf v; } u;
  u.f[0] = lo; u.f[1] = hi;
  return u.v;
}

// A-fragment (16x32 bf16, row-major source).  Layout: lane m = lane%16,
// elements 0..15 => K = half*16 + j (contiguous, matches B fragments).
static __device__ __forceinline__ v16bf load_a_frag(const __bf16* A, int lda,
                                                    int r0, int k0, int l16, int hl) {
  const __bf16* p = A + (size_t)(r0 + l16) * lda + k0 + hl * 16;
  return frag_from2(*(const f32x4*)p, *(const f32x4*)(p + 8));
}

// B-fragment (32x16) from a matrix stored as src[n][k] row-major (i.e. the
// GEMM "B transposed" in memory).  Lane holds fixed n = lane%16 and 16
// contiguous k values starting at k0 + half*16 -> one 32-byte read.
static __device__ __forceinline__ v16bf load_bT_frag(const __bf16* Bt, int ldb,
                                                     int n0, int k0, int l16, int hl) {
  const __bf16* p = Bt + (size_t)(n0 + l16) * ldb + k0 + hl * 16;
  return frag_from2(*(const f32x4*)p, *(const f32x4*)(p + 8));
}

// B-fragment from pre-packed weights: lane's 16 elements are contiguous.
static __device__ __forceinline__ v16bf load_packed_frag(const __bf16* P,
                                                         size_t tile, int lane) {
  const __bf16* p = P + (tile * 32 + (size_t)lane) * 16;
  return frag_from2(*(const f32x4*)p, *(const f32x4*)(p + 8));
}

// ---------------------------------------------------------------------------
// 1) Pack an f32 KxN weight into bf16 WMMA-B-fragment tiles, zero padded in N.
//    grid.x = Ntiles*Kt, block = 32 (one wave packs one 32x16 tile).
// ---------------------------------------------------------------------------
__global__ void k_pack_w(const float* __restrict__ W, __bf16* __restrict__ P,
                         int N, int Kt) {
  int tile  = blockIdx.x;              // tile = ntile*Kt + ktile
  int ktile = tile % Kt;
  int ntile = tile / Kt;
  int lane = threadIdx.x & 31, l16 = lane & 15, hl = lane >> 4;
  int n = ntile * 16 + l16;
  alignas(16) __bf16 tmp[16];
#pragma unroll
  for (int j = 0; j < 16; ++j) {
    int k = ktile * 32 + hl * 16 + j;
    tmp[j] = (__bf16)((n < N) ? W[(size_t)k * N + n] : 0.0f);
  }
  __bf16* dst = P + ((size_t)tile * 32 + lane) * 16;
  *(f32x4*)dst       = *(const f32x4*)tmp;
  *(f32x4*)(dst + 8) = *(const f32x4*)(tmp + 8);
}

// ---------------------------------------------------------------------------
// 2) x[m] = bf16(tok_emb[inputs[m]] + pos_emb[m % SEQ])
// ---------------------------------------------------------------------------
__global__ void k_embed(const int* __restrict__ inputs,
                        const float* __restrict__ tok_emb,
                        const float* __restrict__ pos_emb,
                        __bf16* __restrict__ x) {
  int m = blockIdx.x;
  int t = m % SEQ;
  int tok = inputs[m];
  for (int e = threadIdx.x; e < EMB; e += blockDim.x)
    x[(size_t)m * EMB + e] =
        (__bf16)(tok_emb[(size_t)tok * EMB + e] + pos_emb[(size_t)t * EMB + e]);
}

// ---------------------------------------------------------------------------
// 3) QKV: C = x @ W.  grid (16, 32, 3), block 256 (8 waves).
//    Each wave: 2 M-tiles x 4 N-tiles (B fragments reused across M).
//    z=0 -> q (row-major bf16), z=1 -> k (row-major), z=2 -> vT (transposed).
// ---------------------------------------------------------------------------
__global__ void k_qkv(const __bf16* __restrict__ x,
                      const __bf16* __restrict__ Wq_p,
                      const __bf16* __restrict__ Wk_p,
                      const __bf16* __restrict__ Wv_p,
                      __bf16* __restrict__ q, __bf16* __restrict__ k,
                      __bf16* __restrict__ vT) {
  int wave = threadIdx.x >> 5, lane = threadIdx.x & 31;
  int l16 = lane & 15, hl = lane >> 4;
  int mt0 = (blockIdx.y * 8 + wave) * 2;
  int nt0 = blockIdx.x * 4;
  const __bf16* Wp = (blockIdx.z == 0) ? Wq_p : (blockIdx.z == 1) ? Wk_p : Wv_p;

  v8f c[2][4] = {};
  for (int kt = 0; kt < KT_E; ++kt) {
    v16bf a0 = load_a_frag(x, EMB, mt0 * 16, kt * 32, l16, hl);
    v16bf a1 = load_a_frag(x, EMB, (mt0 + 1) * 16, kt * 32, l16, hl);
#pragma unroll
    for (int j = 0; j < 4; ++j) {
      v16bf bf = load_packed_frag(Wp, (size_t)(nt0 + j) * KT_E + kt, lane);
      c[0][j] = WMMA_BF16(a0, bf, c[0][j]);
      c[1][j] = WMMA_BF16(a1, bf, c[1][j]);
    }
  }

  if (blockIdx.z < 2) {
    __bf16* out = (blockIdx.z == 0) ? q : k;
#pragma unroll
    for (int i = 0; i < 2; ++i) {
#pragma unroll
      for (int j = 0; j < 4; ++j) {
        int n = (nt0 + j) * 16 + l16;
#pragma unroll
        for (int g = 0; g < 8; ++g) {
          int m = (mt0 + i) * 16 + g + hl * 8;
          out[(size_t)m * HEAD + n] = (__bf16)c[i][j][g];
        }
      }
    }
  } else {
    // transposed store: lane holds fixed n=h and 8 contiguous t values
#pragma unroll
    for (int i = 0; i < 2; ++i) {
      int b  = ((mt0 + i) * 16) / SEQ;
      int t0 = ((mt0 + i) * 16) % SEQ + hl * 8;
#pragma unroll
      for (int j = 0; j < 4; ++j) {
        int n = (nt0 + j) * 16 + l16;
        alignas(16) __bf16 tmp[8];
#pragma unroll
        for (int g = 0; g < 8; ++g) tmp[g] = (__bf16)c[i][j][g];
        __bf16* dst = vT + ((size_t)b * HEAD + n) * SEQ + t0;
        *(f32x4*)dst = *(const f32x4*)tmp;
      }
    }
  }
}

// ---------------------------------------------------------------------------
// 4) Attention for one (batch, 16-row t-tile).  grid (128, 4), block 256.
//    Phase 1: scores = q k^T / sqrt(H) into 64KB LDS (bf16); fully-masked
//             s-tiles skip their K-loop entirely (uniform per wave).
//    Phase 2: exact softmax (shfl-xor lane reductions, 2 rows per wave).
//    Phase 3: attn = w @ v, iterating only s-tiles at/below the diagonal.
// ---------------------------------------------------------------------------
__global__ void k_attn(const __bf16* __restrict__ q, const __bf16* __restrict__ k,
                       const __bf16* __restrict__ vT, __bf16* __restrict__ attn) {
  __shared__ __bf16 S[16 * SEQ];       // 64 KB of the 320 KB/WGP LDS
  int wave = threadIdx.x >> 5, lane = threadIdx.x & 31;
  int l16 = lane & 15, hl = lane >> 4;
  int ttile = blockIdx.x, b = blockIdx.y;
  int tmax = ttile * 16 + 15;
  const float scale = 0.03125f;        // 1/sqrt(1024)
  const __bf16* qb = q + (size_t)b * SEQ * HEAD;
  const __bf16* kb = k + (size_t)b * SEQ * HEAD;
  const __bf16* vb = vT + (size_t)b * HEAD * SEQ;

  // ---- phase 1: score rows [ttile*16, +16), two s-tiles per iteration ----
  for (int sp = wave; sp < SEQ / 32; sp += 8) {
    int st0 = sp * 2;                  // s-tiles st0, st0+1
    if (st0 * 16 > tmax) {
      // both tiles entirely above the diagonal -> weight 0, skip compute
#pragma unroll
      for (int g = 0; g < 8; ++g) {
        int m = g + hl * 8;
        S[m * SEQ + st0 * 16 + l16]      = (__bf16)(-1e30f);
        S[m * SEQ + st0 * 16 + 16 + l16] = (__bf16)(-1e30f);
      }
      continue;
    }
    v8f c0 = {}, c1 = {};
    for (int kt = 0; kt < HEAD / 32; ++kt) {
      v16bf a = load_a_frag(qb, HEAD, ttile * 16, kt * 32, l16, hl);
      c0 = WMMA_BF16(a, load_bT_frag(kb, HEAD, st0 * 16, kt * 32, l16, hl), c0);
      c1 = WMMA_BF16(a, load_bT_frag(kb, HEAD, st0 * 16 + 16, kt * 32, l16, hl), c1);
    }
#pragma unroll
    for (int g = 0; g < 8; ++g) {
      int m = g + hl * 8;
      int t = ttile * 16 + m;
      int s0 = st0 * 16 + l16;
      float v0 = c0[g] * scale;
      float v1 = c1[g] * scale;
      if (s0 > t)      v0 = -1e30f;
      if (s0 + 16 > t) v1 = -1e30f;
      S[m * SEQ + s0]      = (__bf16)v0;
      S[m * SEQ + s0 + 16] = (__bf16)v1;
    }
  }
  __syncthreads();

  // ---- phase 2: softmax, wave w handles rows 2w (lanes 0-15), 2w+1 ----
  {
    int r = 2 * wave + hl;
    float mx = -1e30f;
    for (int i = l16; i < SEQ; i += 16) mx = fmaxf(mx, (float)S[r * SEQ + i]);
#pragma unroll
    for (int msk = 1; msk < 16; msk <<= 1) mx = fmaxf(mx, __shfl_xor(mx, msk, 32));
    float sum = 0.0f;
    for (int i = l16; i < SEQ; i += 16) sum += __expf((float)S[r * SEQ + i] - mx);
#pragma unroll
    for (int msk = 1; msk < 16; msk <<= 1) sum += __shfl_xor(sum, msk, 32);
    float inv = 1.0f / sum;
    for (int i = l16; i < SEQ; i += 16)
      S[r * SEQ + i] = (__bf16)(__expf((float)S[r * SEQ + i] - mx) * inv);
  }
  __syncthreads();

  // ---- phase 3: attn = w @ v over s-tiles at/below the diagonal ----
  int nst = tmax / 32 + 1;             // 32-wide K-steps with nonzero weight
  for (int hp = wave; hp < HEAD / 32; hp += 8) {
    int ht0 = hp * 2;                  // h-tiles ht0, ht0+1
    v8f c0 = {}, c1 = {};
    for (int st = 0; st < nst; ++st) {
      v16bf a = load_a_frag(S, SEQ, 0, st * 32, l16, hl);
      c0 = WMMA_BF16(a, load_bT_frag(vb, SEQ, ht0 * 16, st * 32, l16, hl), c0);
      c1 = WMMA_BF16(a, load_bT_frag(vb, SEQ, (ht0 + 1) * 16, st * 32, l16, hl), c1);
    }
#pragma unroll
    for (int g = 0; g < 8; ++g) {
      int t = ttile * 16 + g + hl * 8;
      attn[((size_t)b * SEQ + t) * HEAD + ht0 * 16 + l16]      = (__bf16)c0[g];
      attn[((size_t)b * SEQ + t) * HEAD + (ht0 + 1) * 16 + l16] = (__bf16)c1[g];
    }
  }
}

// ---------------------------------------------------------------------------
// 5) LM head: logits = attn @ W_lm + b.  grid (786, 32), block 256.
//    Each wave: 2 M-tiles x 4 N-tiles, K loop of 32 x 8 WMMAs.
// ---------------------------------------------------------------------------
__global__ void k_lmhead(const __bf16* __restrict__ attn,
                         const __bf16* __restrict__ Wp,
                         const float* __restrict__ bias,
                         float* __restrict__ logits) {
  int wave = threadIdx.x >> 5, lane = threadIdx.x & 31;
  int l16 = lane & 15, hl = lane >> 4;
  int mt0 = (blockIdx.y * 8 + wave) * 2;
  int nt0 = blockIdx.x * 4;            // padded weights: always 4 tiles valid

  v8f c[2][4] = {};
  for (int kt = 0; kt < KT_E; ++kt) {
    v16bf a0 = load_a_frag(attn, HEAD, mt0 * 16, kt * 32, l16, hl);
    v16bf a1 = load_a_frag(attn, HEAD, (mt0 + 1) * 16, kt * 32, l16, hl);
#pragma unroll
    for (int j = 0; j < 4; ++j) {
      v16bf bf = load_packed_frag(Wp, (size_t)(nt0 + j) * KT_E + kt, lane);
      c[0][j] = WMMA_BF16(a0, bf, c[0][j]);
      c[1][j] = WMMA_BF16(a1, bf, c[1][j]);
    }
  }
#pragma unroll
  for (int j = 0; j < 4; ++j) {
    int n = (nt0 + j) * 16 + l16;
    if (n < VOCAB) {
      float bv = bias[n];
#pragma unroll
      for (int i = 0; i < 2; ++i) {
#pragma unroll
        for (int g = 0; g < 8; ++g) {
          int m = (mt0 + i) * 16 + g + hl * 8;
          logits[(size_t)m * VOCAB + n] = c[i][j][g] + bv;
        }
      }
    }
  }
}

// ---------------------------------------------------------------------------
// 6) Per-row log-sum-exp + NLL (deterministic fixed-tree reductions).
// ---------------------------------------------------------------------------
__global__ void k_rowstats(const float* __restrict__ logits,
                           const int* __restrict__ targets,
                           float* __restrict__ nll) {
  __shared__ float red[256];
  int r = blockIdx.x, tid = threadIdx.x;
  const float* p = logits + (size_t)r * VOCAB;
  float mx = -3.0e38f;
  for (int i = tid; i < VOCAB; i += 256) mx = fmaxf(mx, p[i]);
  red[tid] = mx; __syncthreads();
  for (int o = 128; o > 0; o >>= 1) {
    if (tid < o) red[tid] = fmaxf(red[tid], red[tid + o]);
    __syncthreads();
  }
  mx = red[0]; __syncthreads();
  float s = 0.0f;
  for (int i = tid; i < VOCAB; i += 256) s += __expf(p[i] - mx);
  red[tid] = s; __syncthreads();
  for (int o = 128; o > 0; o >>= 1) {
    if (tid < o) red[tid] += red[tid + o];
    __syncthreads();
  }
  if (tid == 0) {
    float lse = mx + __logf(red[0]);
    nll[r] = lse - p[targets[r]];
  }
}

// ---------------------------------------------------------------------------
// 7) loss = mean(nll)
// ---------------------------------------------------------------------------
__global__ void k_loss(const float* __restrict__ nll, float* __restrict__ out) {
  __shared__ float red[256];
  int tid = threadIdx.x;
  float s = 0.0f;
  for (int i = tid; i < MROWS; i += 256) s += nll[i];
  red[tid] = s; __syncthreads();
  for (int o = 128; o > 0; o >>= 1) {
    if (tid < o) red[tid] += red[tid + o];
    __syncthreads();
  }
  if (tid == 0) out[0] = red[0] / (float)MROWS;
}

// ---------------------------------------------------------------------------
extern "C" void kernel_launch(void* const* d_in, const int* in_sizes, int n_in,
                              void* d_out, int out_size, void* d_ws, size_t ws_size,
                              hipStream_t stream) {
  const int*   inputs  = (const int*)d_in[0];
  const int*   targets = (const int*)d_in[1];
  const float* tok_emb = (const float*)d_in[2];
  const float* pos_emb = (const float*)d_in[3];
  const float* Wk      = (const float*)d_in[4];   // note: Wk before Wq in dict
  const float* Wq      = (const float*)d_in[5];
  const float* Wv      = (const float*)d_in[6];
  const float* W_lm    = (const float*)d_in[7];
  const float* b_lm    = (const float*)d_in[8];

  float* logits = (float*)d_out;
  float* loss   = logits + (size_t)MROWS * VOCAB;

  char* ws = (char*)d_ws;
  size_t off = 0;
  auto alloc = [&](size_t bytes) {
    void* p = ws + off;
    off += (bytes + 255) & ~(size_t)255;
    return p;
  };
  __bf16* x      = (__bf16*)alloc((size_t)MROWS * EMB * 2);       // 16.8 MB
  __bf16* qb     = (__bf16*)alloc((size_t)MROWS * HEAD * 2);      // 16.8 MB
  __bf16* kb     = (__bf16*)alloc((size_t)MROWS * HEAD * 2);      // 16.8 MB
  __bf16* vTb    = (__bf16*)alloc((size_t)MROWS * HEAD * 2);      // 16.8 MB
  __bf16* attn   = (__bf16*)alloc((size_t)MROWS * HEAD * 2);      // 16.8 MB
  __bf16* Wq_p   = (__bf16*)alloc((size_t)(HEAD / 16) * KT_E * 512 * 2);  // 2 MB
  __bf16* Wk_p   = (__bf16*)alloc((size_t)(HEAD / 16) * KT_E * 512 * 2);
  __bf16* Wv_p   = (__bf16*)alloc((size_t)(HEAD / 16) * KT_E * 512 * 2);
  __bf16* Wlm_p  = (__bf16*)alloc((size_t)NT_LM_PAD * KT_E * 512 * 2);    // 103 MB
  float*  nll    = (float*)alloc((size_t)MROWS * 4);

  // 1) pack weights into WMMA B-fragment layout (bf16, zero padded in N)
  k_pack_w<<<dim3((HEAD / 16) * KT_E), 32, 0, stream>>>(Wq, Wq_p, HEAD, KT_E);
  k_pack_w<<<dim3((HEAD / 16) * KT_E), 32, 0, stream>>>(Wk, Wk_p, HEAD, KT_E);
  k_pack_w<<<dim3((HEAD / 16) * KT_E), 32, 0, stream>>>(Wv, Wv_p, HEAD, KT_E);
  k_pack_w<<<dim3(NT_LM_PAD * KT_E), 32, 0, stream>>>(W_lm, Wlm_p, VOCAB, KT_E);

  // 2) embedding
  k_embed<<<dim3(MROWS), 256, 0, stream>>>(inputs, tok_emb, pos_emb, x);

  // 3) q, k, vT projections (z selects the weight / output / store mode)
  k_qkv<<<dim3(HEAD / 64, MROWS / 256, 3), 256, 0, stream>>>(
      x, Wq_p, Wk_p, Wv_p, qb, kb, vTb);

  // 4) causal attention
  k_attn<<<dim3(SEQ / 16, BATCH), 256, 0, stream>>>(qb, kb, vTb, attn);

  // 5) LM head -> f32 logits in d_out
  k_lmhead<<<dim3(NT_LM_PAD / 4, MROWS / 256), 256, 0, stream>>>(
      attn, Wlm_p, b_lm, logits);

  // 6) per-row LSE + NLL, 7) mean loss
  k_rowstats<<<dim3(MROWS), 256, 0, stream>>>(logits, targets, nll);
  k_loss<<<dim3(1), 256, 0, stream>>>(nll, loss);
}
